// Attention_68642167325265
// MI455X (gfx1250) — matrix operand order repaired and checked by reading, not compiled, not run on hardware
//
#include <hip/hip_runtime.h>
#include <hip/hip_bf16.h>

// ---------------------------------------------------------------------------
// Fused attention block for gfx1250 (MI455X), bf16 WMMA with f32 accumulation.
//   stage 0: one-shot fp32 -> bf16 conversion of x, qkv_w, proj_w
//   stage 1: x @ qkv_w^T -> Q,K (bf16 [B,H,N,64]) and V^T (bf16 [B,H,64,N])
//   stage 2: flash-attention per (b,h) -> AO (bf16 [B*N, C])
//   stage 3: AO @ proj_w^T + proj_b -> out (f32 [B*N, C])
// GEMM hot loops are software-pipelined with the last K-step peeled, so the
// next tile's fragments always load unconditionally into fresh SSA registers
// (no zero-fill movs, no WMMA WAR hazard NOPs), and read pure bf16.
// ---------------------------------------------------------------------------

typedef __bf16 v16bf __attribute__((ext_vector_type(16)));
typedef __bf16 v8bf  __attribute__((ext_vector_type(8)));
typedef float  v8f   __attribute__((ext_vector_type(8)));

#define DIM   1024
#define NHEAD 16
#define HDIM  64
#define BATCH 8
#define SEQ   1024
#define MTOT  (BATCH * SEQ)      // 8192 rows
#define QKV_N (3 * DIM)          // 3072

static __device__ __forceinline__ v8f wmma_bf16(v16bf a, v16bf b, v8f c) {
  // D = A(16x32 bf16) * B(32x16 bf16) + C(16x16 f32)
  return __builtin_amdgcn_wmma_f32_16x16x32_bf16(
      /*neg_a=*/false, a, /*neg_b=*/false, b,
      /*c_mod=*/(short)0, c, /*reuse_a=*/false, /*reuse_b=*/false);
}

// A fragment (16x32 bf16). Lane layout: m = lane%16, half = lane/16;
// elems [0..15] = A[m][16*half + 0..15].
static __device__ __forceinline__ v16bf load_a_bf16(const __bf16* p, int lh) {
  v16bf r;
  const __bf16* p0 = p + 16 * lh;
  const __bf16* p1 = p + 16 * lh + 8;
#pragma unroll
  for (int i = 0; i < 8; ++i) { r[i] = p0[i]; r[8 + i] = p1[i]; }
  return r;
}

// B fragment (32x16 bf16): lane holds column n = lane%16, K values
// 16*(lane/16)+0..15 -> 16 contiguous elements at the caller-supplied pointer.
static __device__ __forceinline__ v16bf load_bT_bf16(const __bf16* p) {
  v16bf r;
#pragma unroll
  for (int i = 0; i < 16; ++i) r[i] = p[i];
  return r;
}

// Issue the 8 WMMAs of one 32x64 wave tile K-step.
static __device__ __forceinline__ void wmma_step(
    v8f acc[2][4], v16bf a0, v16bf a1, v16bf b0, v16bf b1, v16bf b2, v16bf b3) {
  acc[0][0] = wmma_bf16(a0, b0, acc[0][0]);
  acc[1][0] = wmma_bf16(a1, b0, acc[1][0]);
  acc[0][1] = wmma_bf16(a0, b1, acc[0][1]);
  acc[1][1] = wmma_bf16(a1, b1, acc[1][1]);
  acc[0][2] = wmma_bf16(a0, b2, acc[0][2]);
  acc[1][2] = wmma_bf16(a1, b2, acc[1][2]);
  acc[0][3] = wmma_bf16(a0, b3, acc[0][3]);
  acc[1][3] = wmma_bf16(a1, b3, acc[1][3]);
}

// ---------------------------------------------------------------------------
// Kernel 0: one-shot fp32 -> bf16 (8 elements per thread, 32B in / 16B out)
// ---------------------------------------------------------------------------
__global__ __launch_bounds__(256) void cvt_bf16_kernel(
    const float* __restrict__ src, __bf16* __restrict__ dst, int n8) {
  int i = blockIdx.x * 256 + threadIdx.x;
  if (i < n8) {
    const float4* s = (const float4*)src;
    float4 lo = s[2 * i], hi = s[2 * i + 1];
    v8bf pk;
    pk[0] = (__bf16)lo.x; pk[1] = (__bf16)lo.y;
    pk[2] = (__bf16)lo.z; pk[3] = (__bf16)lo.w;
    pk[4] = (__bf16)hi.x; pk[5] = (__bf16)hi.y;
    pk[6] = (__bf16)hi.z; pk[7] = (__bf16)hi.w;
    ((v8bf*)dst)[i] = pk;
  }
}

// ---------------------------------------------------------------------------
// Kernel 1: QKV GEMM.  C[8192,3072] = xb[8192,1024] * wb[3072,1024]^T (bf16)
// WG tile 64x256 (8 waves, 2x4), wave tile 32x64 (2x4 accumulators).
// Epilogue: Q,K row-major [B,H,N,64]; V transposed [B,H,64,N] with packed
// 16-byte stores (each wave's 64 cols sit in exactly one (t, head) slice).
// ---------------------------------------------------------------------------
__global__ __launch_bounds__(256) void qkv_gemm_kernel(
    const __bf16* __restrict__ x, const __bf16* __restrict__ w,
    __bf16* __restrict__ Q, __bf16* __restrict__ Kd, __bf16* __restrict__ V) {
  const int lane = threadIdx.x & 31;
  const int wid  = threadIdx.x >> 5;
  const int l16  = lane & 15, lh = lane >> 4;
  const int wm = wid & 1, wn = wid >> 1;
  const int row0 = blockIdx.x * 64 + wm * 32;
  const int col0 = blockIdx.y * 256 + wn * 64;

  v8f acc[2][4] = {};

  const __bf16* arow0 = x + (size_t)(row0 + l16) * DIM;
  const __bf16* arow1 = x + (size_t)(row0 + 16 + l16) * DIM;
  const __bf16* brow[4];
#pragma unroll
  for (int j = 0; j < 4; ++j)
    brow[j] = w + (size_t)(col0 + j * 16 + l16) * DIM + 16 * lh;

  // Software pipeline with peeled final step: loads are always unconditional.
  v16bf a0 = load_a_bf16(arow0, lh);
  v16bf a1 = load_a_bf16(arow1, lh);
  v16bf bf0 = load_bT_bf16(brow[0]);
  v16bf bf1 = load_bT_bf16(brow[1]);
  v16bf bf2 = load_bT_bf16(brow[2]);
  v16bf bf3 = load_bT_bf16(brow[3]);

#pragma unroll 2
  for (int kb = 0; kb < DIM - 32; kb += 32) {
    __builtin_prefetch(arow0 + kb + 288, 0, 1);
    v16bf na0 = load_a_bf16(arow0 + kb + 32, lh);
    v16bf na1 = load_a_bf16(arow1 + kb + 32, lh);
    v16bf nb0 = load_bT_bf16(brow[0] + kb + 32);
    v16bf nb1 = load_bT_bf16(brow[1] + kb + 32);
    v16bf nb2 = load_bT_bf16(brow[2] + kb + 32);
    v16bf nb3 = load_bT_bf16(brow[3] + kb + 32);
    wmma_step(acc, a0, a1, bf0, bf1, bf2, bf3);
    a0 = na0; a1 = na1;
    bf0 = nb0; bf1 = nb1; bf2 = nb2; bf3 = nb3;
  }
  wmma_step(acc, a0, a1, bf0, bf1, bf2, bf3);   // peeled last K-step

  // ---- epilogue: (t, head, batch) are wave-uniform ----
  const int tW  = col0 >> 10;               // 0=Q 1=K 2=V
  const int hhW = (col0 & 1023) >> 6;       // head
  const int bb  = row0 >> 10;               // batch
  const int nsBase = row0 & 1023;           // seq base of wave tile

  if (tW == 2) {
    // V transposed: VT[e][ns], contiguous across the 8 rows each lane holds.
    __bf16* vt = V + (size_t)(bb * NHEAD + hhW) * HDIM * SEQ;
#pragma unroll
    for (int mi = 0; mi < 2; ++mi) {
#pragma unroll
      for (int j = 0; j < 4; ++j) {
        int e   = j * 16 + l16;
        int ns0 = nsBase + mi * 16 + 8 * lh;
        v8bf pk;
#pragma unroll
        for (int r = 0; r < 8; ++r) pk[r] = (__bf16)acc[mi][j][r];
        *(v8bf*)(vt + (size_t)e * SEQ + ns0) = pk;   // 16B aligned (ns0 % 8 == 0)
      }
    }
  } else {
    __bf16* dst = (tW == 0 ? Q : Kd) + (size_t)(bb * NHEAD + hhW) * SEQ * HDIM;
#pragma unroll
    for (int mi = 0; mi < 2; ++mi) {
#pragma unroll
      for (int j = 0; j < 4; ++j) {
#pragma unroll
        for (int r = 0; r < 8; ++r) {
          int ns = nsBase + mi * 16 + r + 8 * lh;
          int e  = j * 16 + l16;
          dst[(size_t)ns * HDIM + e] = (__bf16)acc[mi][j][r];
        }
      }
    }
  }
}

// ---------------------------------------------------------------------------
// Kernel 2: flash attention, one (b,h) slice per 8 blocks, one 16-query tile
// per wave, online softmax over 32-key chunks. P transposed via per-wave LDS.
// V is consumed transposed -> contiguous B-fragment loads.
// ---------------------------------------------------------------------------
__global__ __launch_bounds__(256) void attn_kernel(
    const __bf16* __restrict__ Q, const __bf16* __restrict__ K,
    const __bf16* __restrict__ V, __bf16* __restrict__ AO) {
  __shared__ __bf16 Pl[8][16][32];   // per-wave 16x32 P tile (1 KB each)

  const int lane = threadIdx.x & 31;
  const int wid  = threadIdx.x >> 5;
  const int l16  = lane & 15, lh = lane >> 4;
  const int bh   = blockIdx.x >> 3;      // 0..127 = b*16 + h
  const int qblk = blockIdx.x & 7;
  const int b = bh >> 4, h = bh & 15;

  const __bf16* Qh  = Q + (size_t)bh * SEQ * HDIM;
  const __bf16* Kh  = K + (size_t)bh * SEQ * HDIM;
  const __bf16* VTh = V + (size_t)bh * HDIM * SEQ;   // [64][1024]

  const int q0 = qblk * 128 + wid * 16;  // 16-query tile base

  // Q A-fragments for hd chunks [0,32) and [32,64) -- loaded once.
  v16bf aq0 = load_a_bf16(Qh + (size_t)(q0 + l16) * HDIM, lh);
  v16bf aq1 = load_a_bf16(Qh + (size_t)(q0 + l16) * HDIM + 32, lh);

  float mrow[8], srow[8];
#pragma unroll
  for (int r = 0; r < 8; ++r) { mrow[r] = -3.0e38f; srow[r] = 0.0f; }
  v8f o[4] = {};

  const float scale = 0.125f;  // 64^-0.5

  for (int kt = 0; kt < SEQ; kt += 32) {
    // --- S = scale * Q K^T for keys [kt, kt+32) ---
    const __bf16* kr0 = Kh + (size_t)(kt + l16) * HDIM + 16 * lh;       // keys kt+0..15
    const __bf16* kr1 = Kh + (size_t)(kt + 16 + l16) * HDIM + 16 * lh;  // keys kt+16..31
    v16bf bk0a = load_bT_bf16(kr0);
    v16bf bk0b = load_bT_bf16(kr0 + 32);
    v16bf bk1a = load_bT_bf16(kr1);
    v16bf bk1b = load_bT_bf16(kr1 + 32);
    v8f s0 = {}, s1 = {};
    s0 = wmma_bf16(aq0, bk0a, s0);
    s0 = wmma_bf16(aq1, bk0b, s0);
    s1 = wmma_bf16(aq0, bk1a, s1);
    s1 = wmma_bf16(aq1, bk1b, s1);

    // V B-fragments (contiguous from V^T); issued early, consumed after softmax.
    v16bf bv0 = load_bT_bf16(VTh + (size_t)(0 * 16 + l16) * SEQ + kt + 16 * lh);
    v16bf bv1 = load_bT_bf16(VTh + (size_t)(1 * 16 + l16) * SEQ + kt + 16 * lh);
    v16bf bv2 = load_bT_bf16(VTh + (size_t)(2 * 16 + l16) * SEQ + kt + 16 * lh);
    v16bf bv3 = load_bT_bf16(VTh + (size_t)(3 * 16 + l16) * SEQ + kt + 16 * lh);

    // --- online softmax update (rows live in lane half-groups of 16) ---
    float alpha[8];
#pragma unroll
    for (int r = 0; r < 8; ++r) {
      float x0 = s0[r] * scale, x1 = s1[r] * scale;
      float mx = fmaxf(x0, x1);
#pragma unroll
      for (int off = 1; off < 16; off <<= 1) mx = fmaxf(mx, __shfl_xor(mx, off, 32));
      float mnew = fmaxf(mrow[r], mx);
      float al = __expf(mrow[r] - mnew);
      float p0 = __expf(x0 - mnew);
      float p1 = __expf(x1 - mnew);
      float ps = p0 + p1;
#pragma unroll
      for (int off = 1; off < 16; off <<= 1) ps += __shfl_xor(ps, off, 32);
      srow[r] = srow[r] * al + ps;
      mrow[r] = mnew;
      alpha[r] = al;
      // Stash P in C-layout into per-wave LDS tile.
      Pl[wid][r + 8 * lh][l16]      = (__bf16)p0;
      Pl[wid][r + 8 * lh][16 + l16] = (__bf16)p1;
    }

    // rescale running accumulator
#pragma unroll
    for (int t = 0; t < 4; ++t)
#pragma unroll
      for (int r = 0; r < 8; ++r) o[t][r] *= alpha[r];

    // Read P back in A-layout (16x32). DScnt waits inserted by compiler.
    v16bf ap;
#pragma unroll
    for (int i = 0; i < 8; ++i) {
      ap[i]     = Pl[wid][l16][16 * lh + i];
      ap[8 + i] = Pl[wid][l16][16 * lh + 8 + i];
    }

    // --- O += P * V ---
    o[0] = wmma_bf16(ap, bv0, o[0]);
    o[1] = wmma_bf16(ap, bv1, o[1]);
    o[2] = wmma_bf16(ap, bv2, o[2]);
    o[3] = wmma_bf16(ap, bv3, o[3]);
  }

  // Epilogue: normalize and scatter to AO[B*N, C] at col h*64 + e.
  float inv[8];
#pragma unroll
  for (int r = 0; r < 8; ++r) inv[r] = 1.0f / srow[r];
#pragma unroll
  for (int t = 0; t < 4; ++t) {
#pragma unroll
    for (int r = 0; r < 8; ++r) {
      int qq = q0 + r + 8 * lh;
      float val = o[t][r] * inv[r];
      AO[((size_t)(b * SEQ + qq)) * DIM + h * HDIM + t * 16 + l16] = (__bf16)val;
    }
  }
}

// ---------------------------------------------------------------------------
// Kernel 3: output projection.  out[8192,1024] = AO * wb^T + proj_b  (bf16 in)
// Same software-pipelined structure as kernel 1 (peeled final K-step).
// ---------------------------------------------------------------------------
__global__ __launch_bounds__(256) void proj_gemm_kernel(
    const __bf16* __restrict__ A, const __bf16* __restrict__ w,
    const float* __restrict__ bias, float* __restrict__ out) {
  const int lane = threadIdx.x & 31;
  const int wid  = threadIdx.x >> 5;
  const int l16  = lane & 15, lh = lane >> 4;
  const int wm = wid & 1, wn = wid >> 1;
  const int row0 = blockIdx.x * 64 + wm * 32;
  const int col0 = blockIdx.y * 256 + wn * 64;

  v8f acc[2][4] = {};

  const __bf16* arow0 = A + (size_t)(row0 + l16) * DIM;
  const __bf16* arow1 = A + (size_t)(row0 + 16 + l16) * DIM;
  const __bf16* brow[4];
#pragma unroll
  for (int j = 0; j < 4; ++j)
    brow[j] = w + (size_t)(col0 + j * 16 + l16) * DIM + 16 * lh;

  v16bf a0 = load_a_bf16(arow0, lh);
  v16bf a1 = load_a_bf16(arow1, lh);
  v16bf bf0 = load_bT_bf16(brow[0]);
  v16bf bf1 = load_bT_bf16(brow[1]);
  v16bf bf2 = load_bT_bf16(brow[2]);
  v16bf bf3 = load_bT_bf16(brow[3]);

#pragma unroll 2
  for (int kb = 0; kb < DIM - 32; kb += 32) {
    __builtin_prefetch(arow0 + kb + 288, 0, 1);
    v16bf na0 = load_a_bf16(arow0 + kb + 32, lh);
    v16bf na1 = load_a_bf16(arow1 + kb + 32, lh);
    v16bf nb0 = load_bT_bf16(brow[0] + kb + 32);
    v16bf nb1 = load_bT_bf16(brow[1] + kb + 32);
    v16bf nb2 = load_bT_bf16(brow[2] + kb + 32);
    v16bf nb3 = load_bT_bf16(brow[3] + kb + 32);
    wmma_step(acc, a0, a1, bf0, bf1, bf2, bf3);
    a0 = na0; a1 = na1;
    bf0 = nb0; bf1 = nb1; bf2 = nb2; bf3 = nb3;
  }
  wmma_step(acc, a0, a1, bf0, bf1, bf2, bf3);   // peeled last K-step

#pragma unroll
  for (int mi = 0; mi < 2; ++mi) {
#pragma unroll
    for (int j = 0; j < 4; ++j) {
#pragma unroll
      for (int r = 0; r < 8; ++r) {
        int m = row0 + mi * 16 + r + 8 * lh;
        int d = col0 + j * 16 + l16;
        out[(size_t)m * DIM + d] = acc[mi][j][r] + bias[d];
      }
    }
  }
}

// ---------------------------------------------------------------------------
// Host launcher. Workspace layout (bf16 elements):
//   Q [0,8M)  K [8M,16M)  V^T [16M,24M)  AO [24M,32M)
//   xb [32M,40M)  wqkvb [40M,43M)  wprojb [43M,44M)      = 88 MB total
// ---------------------------------------------------------------------------
extern "C" void kernel_launch(void* const* d_in, const int* in_sizes, int n_in,
                              void* d_out, int out_size, void* d_ws, size_t ws_size,
                              hipStream_t stream) {
  const float* x      = (const float*)d_in[0];
  const float* qkv_w  = (const float*)d_in[1];
  const float* proj_w = (const float*)d_in[2];
  const float* proj_b = (const float*)d_in[3];
  float* out = (float*)d_out;
  (void)in_sizes; (void)n_in; (void)out_size; (void)ws_size;

  const size_t QKV_ELEMS = (size_t)MTOT * DIM;        // 8388608
  const size_t XW_ELEMS  = (size_t)MTOT * DIM;        // 8388608
  const size_t WQ_ELEMS  = (size_t)QKV_N * DIM;       // 3145728
  const size_t WP_ELEMS  = (size_t)DIM * DIM;         // 1048576

  __bf16* Q   = (__bf16*)d_ws;
  __bf16* K   = Q + QKV_ELEMS;
  __bf16* V   = K + QKV_ELEMS;   // transposed per head: [B,H,64,1024]
  __bf16* AO  = V + QKV_ELEMS;
  __bf16* xb  = AO + QKV_ELEMS;
  __bf16* wqb = xb + XW_ELEMS;
  __bf16* wpb = wqb + WQ_ELEMS;

  // 0) one-shot bf16 conversions (8 elems/thread)
  cvt_bf16_kernel<<<dim3((int)(XW_ELEMS / 8 / 256)), 256, 0, stream>>>(x, xb, (int)(XW_ELEMS / 8));
  cvt_bf16_kernel<<<dim3((int)(WQ_ELEMS / 8 / 256)), 256, 0, stream>>>(qkv_w, wqb, (int)(WQ_ELEMS / 8));
  cvt_bf16_kernel<<<dim3((int)(WP_ELEMS / 8 / 256)), 256, 0, stream>>>(proj_w, wpb, (int)(WP_ELEMS / 8));

  // 1) QKV projection: grid (8192/64, 3072/256)
  qkv_gemm_kernel<<<dim3(MTOT / 64, QKV_N / 256), 256, 0, stream>>>(
      xb, wqb, Q, K, V);

  // 2) attention: 128 (b,h) slices x 8 query blocks
  attn_kernel<<<dim3(BATCH * NHEAD * (SEQ / 128)), 256, 0, stream>>>(Q, K, V, AO);

  // 3) output projection: grid (8192/64, 1024/256)
  proj_gemm_kernel<<<dim3(MTOT / 64, DIM / 256), 256, 0, stream>>>(
      AO, wpb, proj_b, out);
}
